// Generator_56762287784495
// MI455X (gfx1250) — matrix operand order repaired and checked by reading, not compiled, not run on hardware
//
#include <hip/hip_runtime.h>

typedef __attribute__((ext_vector_type(16))) _Float16 v16h;
typedef __attribute__((ext_vector_type(8)))  _Float16 v8h;
typedef __attribute__((ext_vector_type(8)))  float    v8f;

namespace {

constexpr int kBatch = 4096;
constexpr int kT     = 64;   // time points
constexpr int kHid   = 64;   // hidden size
constexpr int kNoise = 8;
constexpr int kData  = 16;
constexpr int kInitN = 32;

constexpr int TILE_M = 64;   // batch rows per workgroup
constexpr int WAVES  = 8;    // 256 threads = 8 wave32

// LDS row strides (padded to dodge bank conflicts; multiples of 8 f16 = 16B)
constexpr int XS = 104;      // tx buffer: 96 used cols (1 + 64 + pad-to-96)
constexpr int PS = 264;      // 256-wide activation ping/pong buffers
constexpr int FS = 68;       // f32 64-wide state buffers

// Packed-weight fragment sizes (f16 elements). Fragment = 32 lanes x 16 f16 = 512.
constexpr int SZ_32x256  = 1 * 16 * 512;
constexpr int SZ_96x256  = 3 * 16 * 512;   // K=65 padded to 96
constexpr int SZ_256x256 = 8 * 16 * 512;
constexpr int SZ_256x64  = 8 * 4  * 512;
constexpr int SZ_256x512 = 8 * 32 * 512;

constexpr int O_IW0 = 0;
constexpr int O_IW1 = O_IW0 + SZ_32x256;
constexpr int O_IW2 = O_IW1 + SZ_256x256;
constexpr int O_IW3 = O_IW2 + SZ_256x256;
constexpr int O_DW0 = O_IW3 + SZ_256x64;
constexpr int O_DW1 = O_DW0 + SZ_96x256;
constexpr int O_DW2 = O_DW1 + SZ_256x256;
constexpr int O_DW3 = O_DW2 + SZ_256x256;
constexpr int O_GW0 = O_DW3 + SZ_256x64;
constexpr int O_GW1 = O_GW0 + SZ_96x256;
constexpr int O_GW2 = O_GW1 + SZ_256x256;
constexpr int O_GW3 = O_GW2 + SZ_256x256;

constexpr size_t LDS_BYTES =
    (size_t)TILE_M * XS * 2                 // X  (tx, f16)
  + 2 * (size_t)TILE_M * PS * 2             // P0, P1 (f16)
  + 7 * (size_t)TILE_M * FS * 4             // Y, Z, F0, F1, GI, GA, GB (f32)
  + 2 * (size_t)TILE_M * kNoise * 4         // DWA, DWB (f32)
  + (size_t)(kHid * kData + kData) * 4;     // readout W + b (f32)

// ---- fast transcendental helpers (single-instruction paths on gfx1250) ----
__device__ __forceinline__ float fast_rcp(float x) {
#if __has_builtin(__builtin_amdgcn_rcpf)
  return __builtin_amdgcn_rcpf(x);
#else
  return 1.0f / x;
#endif
}
__device__ __forceinline__ float fast_sigmoid(float v) {
  return fast_rcp(1.0f + __expf(-v));
}
__device__ __forceinline__ float fast_tanh(float v) {
#if __has_builtin(__builtin_amdgcn_tanhf)
  return __builtin_amdgcn_tanhf(v);
#else
  // tanh(x) = 1 - 2/(exp(2x)+1)
  return 1.0f - 2.0f * fast_rcp(__expf(2.0f * v) + 1.0f);
#endif
}

// ---------------------------------------------------------------------------
// Pack a row-major f32 weight [K x N] into WMMA B-fragments (f16).
// Fragment (kb, nb): lane L holds column n = nb*16 + (L&15),
// K-values k = kb*32 + ((L>>4)<<4) + i for i in [0,16)  (contiguous K order, same as the A loads).
// Stored lane-major: dst[frag*512 + L*16 + i]. Rows k >= K are zero (K padding).
// ---------------------------------------------------------------------------
__global__ void pack_weights(const float* __restrict__ src,
                             _Float16* __restrict__ dst,
                             int K, int N, int KBT) {
  const int NBT = N >> 4;
  const int total = KBT * NBT * 512;
  for (int idx = blockIdx.x * blockDim.x + threadIdx.x; idx < total;
       idx += gridDim.x * blockDim.x) {
    const int f    = idx >> 9;
    const int r    = idx & 511;
    const int lane = r >> 4;
    const int i    = r & 15;
    const int kb   = f / NBT;
    const int nb   = f - kb * NBT;
    const int n    = nb * 16 + (lane & 15);
    const int k    = kb * 32 + ((lane >> 4) << 4) + i;
    const float v  = (k < K) ? src[(size_t)k * N + n] : 0.0f;
    dst[idx] = (_Float16)v;
  }
}

struct GenParams {
  const float* ts;
  const float* init_noise;
  const float* dW;                 // [T-1][B][NOISE] unit normals
  const float *ib0, *ib1, *ib2, *ib3;   // initial MLP biases
  const float *db0, *db1, *db2, *db3;   // drift MLP biases
  const float *gb0, *gb1, *gb2, *gb3;   // diffusion MLP biases
  const float *rW, *rb;            // readout
  const _Float16* wpk;             // packed weights in workspace
  float* out;                      // [B][T][1+DATA]
};

// ---------------------------------------------------------------------------
// One dense layer: [TILE_M x 32*KBT] (f16, LDS) @ [32*KBT x 16*NBT] (packed f16)
// 8 waves split a grid of 2Mx{1,4}N WMMA-tile quads. At NP=4 (wide layers) each
// wave owns whole quads (one at N=256, two at N=512): per K-step 2 A-frag ds loads + 4 B-frag
// global loads feed 8 back-to-back WMMAs (0.75 memory ops per WMMA).
// MODE 0: LipSwish -> f16 act buffer      MODE 1: linear -> two f32 buffers (y,z)
// MODE 2: tanh -> f32 buffer (f)          MODE 3: tanh, contract with dW_cur and
//                                                 dW_nxt via ds_add_f32 into LDS
// ---------------------------------------------------------------------------
template<int MODE, int KBT, int NBT>
__device__ __forceinline__ void mlp_layer(
    const _Float16* __restrict__ wf, const float* __restrict__ bias,
    const _Float16* ain, int istride,
    _Float16* aout, int ostride,
    float* fa, float* fb,
    const float* dwc, const float* dwn,
    int lane, int wave) {
  const int lm = lane & 15;
  const int lh = lane >> 4;
  constexpr int MP = 2;                        // M tiles per quad
  constexpr int NP = (NBT >= 16) ? 4 : 1;      // N tiles per quad
  constexpr int MQ = (TILE_M / 16) / MP;       // 2
  constexpr int GRID = MQ * (NBT / NP);
#pragma unroll
  for (int t = wave; t < GRID; t += WAVES) {
    const int mq = (t % MQ) * MP;
    const int nq = (t / MQ) * NP;
    v8f acc[MP][NP] = {};
#pragma unroll
    for (int kb = 0; kb < KBT; ++kb) {
      v16h a[MP];
#pragma unroll
      for (int i = 0; i < MP; ++i) {
        // A-fragment (contiguous K order, matching packed B): lane half selects K 16h..16h+15
        const _Float16* arow =
            ain + ((mq + i) * 16 + lm) * istride + kb * 32 + lh * 16;
        v8h alo = *(const v8h*)(arow);        // K = base .. base+7
        v8h ahi = *(const v8h*)(arow + 8);    // K = base+8 .. base+15
        a[i] = __builtin_shufflevector(alo, ahi,
                 0, 1, 2, 3, 4, 5, 6, 7, 8, 9, 10, 11, 12, 13, 14, 15);
      }
#pragma unroll
      for (int j = 0; j < NP; ++j) {
        v16h b = *((const v16h*)wf + ((size_t)kb * NBT + (nq + j)) * 32 + lane);
#pragma unroll
        for (int i = 0; i < MP; ++i) {
          acc[i][j] = __builtin_amdgcn_wmma_f32_16x16x32_f16(
                          false, a[i], false, b, (short)0, acc[i][j],
                          false, false);
        }
      }
    }
#pragma unroll
    for (int j = 0; j < NP; ++j) {
      const int ng  = (nq + j) * 16 + lm;   // global output column
      const float bs = bias[ng];
#pragma unroll
      for (int i = 0; i < MP; ++i) {
#pragma unroll
        for (int r = 0; r < 8; ++r) {
          const int mg = (mq + i) * 16 + lh * 8 + r;   // global row (C layout)
          float v = acc[i][j][r] + bs;
          if (MODE == 0) {
            // LipSwish: 0.909 * x * sigmoid(x)
            aout[mg * ostride + ng] = (_Float16)(0.909f * v * fast_sigmoid(v));
          } else if (MODE == 1) {
            fa[mg * FS + ng] = v;           // y
            fb[mg * FS + ng] = v;           // z
          } else if (MODE == 2) {
            fa[mg * FS + ng] = fast_tanh(v);    // drift f
          } else {
            // diffusion: g = tanh(v); column ng = h*NOISE + nz
            const float g  = fast_tanh(v);
            const int   h  = ng >> 3;
            const int   nz = ng & 7;
            atomicAdd(&fa[mg * FS + h], g * dwc[mg * kNoise + nz]);
            atomicAdd(&fb[mg * FS + h], g * dwn[mg * kNoise + nz]);
          }
        }
      }
    }
  }
}

__device__ __forceinline__ void readout_step(
    const float* Y, float* out, int r0, int n, float tval,
    const float* RWL, const float* RBL, int tid) {
  for (int p = tid; p < TILE_M * kData; p += 256) {
    const int m = p >> 4;
    const int d = p & 15;
    float acc = RBL[d];
#pragma unroll 8
    for (int h = 0; h < kHid; ++h) acc += Y[m * FS + h] * RWL[h * kData + d];
    const size_t base = ((size_t)(r0 + m) * kT + n) * (1 + kData);
    out[base + 1 + d] = acc;
    if (d == 0) out[base] = tval;
  }
}

__global__ __launch_bounds__(256) void sde_generator(GenParams p) {
  extern __shared__ char smem[];
  _Float16* X  = (_Float16*)smem;
  _Float16* P0 = (_Float16*)(smem + (size_t)TILE_M * XS * 2);
  _Float16* P1 = (_Float16*)(smem + (size_t)TILE_M * XS * 2 + (size_t)TILE_M * PS * 2);
  float* Y   = (float*)(smem + (size_t)TILE_M * XS * 2 + 2 * (size_t)TILE_M * PS * 2);
  float* Z   = Y   + TILE_M * FS;
  float* F0  = Z   + TILE_M * FS;
  float* F1  = F0  + TILE_M * FS;
  float* GI  = F1  + TILE_M * FS;   // carried g_prev . dW_n
  float* GA  = GI  + TILE_M * FS;   // g_new . dW_n
  float* GB  = GA  + TILE_M * FS;   // g_new . dW_{n+1}
  float* DWA = GB  + TILE_M * FS;   // [TILE_M][NOISE]
  float* DWB = DWA + TILE_M * kNoise;
  float* RWL = DWB + TILE_M * kNoise;          // readout W [64][16]
  float* RBL = RWL + kHid * kData;             // readout b [16]

  const int tid  = threadIdx.x;
  const int lane = tid & 31;
  const int wave = tid >> 5;
  const int r0   = blockIdx.x * TILE_M;

  const float dt  = p.ts[1] - p.ts[0];
  const float sdt = sqrtf(dt);
  const _Float16* wpk = p.wpk;

  // ---- one-time staging: init noise, tx pad cols, readout weights ----------
  for (int e = tid; e < TILE_M * kInitN; e += 256) {
    const int m = e >> 5, c = e & 31;
    X[m * XS + c] = (_Float16)p.init_noise[(size_t)(r0 + m) * kInitN + c];
  }
  for (int e = tid; e < TILE_M * 31; e += 256) {
    const int m = e / 31, c = e - m * 31;
    X[m * XS + 65 + c] = (_Float16)0.0f;   // K-pad cols 65..95, written once
  }
  for (int e = tid; e < kHid * kData; e += 256) RWL[e] = p.rW[e];
  if (tid < kData) RBL[tid] = p.rb[tid];
  __syncthreads();

  // ---- t = 0: initial MLP (noise -> x0 = y = z) ----------------------------
  mlp_layer<0, 1, 16>(wpk + O_IW0, p.ib0, X,  XS, P0, PS, nullptr, nullptr, nullptr, nullptr, lane, wave); __syncthreads();
  mlp_layer<0, 8, 16>(wpk + O_IW1, p.ib1, P0, PS, P1, PS, nullptr, nullptr, nullptr, nullptr, lane, wave); __syncthreads();
  mlp_layer<0, 8, 16>(wpk + O_IW2, p.ib2, P1, PS, P0, PS, nullptr, nullptr, nullptr, nullptr, lane, wave); __syncthreads();
  mlp_layer<1, 8,  4>(wpk + O_IW3, p.ib3, P0, PS, nullptr, 0, Y, Z, nullptr, nullptr, lane, wave); __syncthreads();

  // readout of x0, build tx(ts[0], x0), load dW_1, zero gdW accumulators
  readout_step(Y, p.out, r0, 0, p.ts[0], RWL, RBL, tid);
  for (int e = tid; e < TILE_M * kHid; e += 256) {
    const int m = e >> 6, h = e & 63;
    X[m * XS + 1 + h] = (_Float16)Z[m * FS + h];
    GI[m * FS + h] = 0.0f;
    GA[m * FS + h] = 0.0f;
  }
  for (int m = tid; m < TILE_M; m += 256) X[m * XS] = (_Float16)p.ts[0];
  for (int e = tid; e < TILE_M * kNoise; e += 256) {
    const int m = e >> 3, j = e & 7;
    DWA[e] = p.dW[(size_t)(r0 + m) * kNoise + j] * sdt;
  }
  __syncthreads();

  // f0 = drift(ts[0], x0)
  mlp_layer<0, 3, 16>(wpk + O_DW0, p.db0, X,  XS, P0, PS, nullptr, nullptr, nullptr, nullptr, lane, wave); __syncthreads();
  mlp_layer<0, 8, 16>(wpk + O_DW1, p.db1, P0, PS, P1, PS, nullptr, nullptr, nullptr, nullptr, lane, wave); __syncthreads();
  mlp_layer<0, 8, 16>(wpk + O_DW2, p.db2, P1, PS, P0, PS, nullptr, nullptr, nullptr, nullptr, lane, wave); __syncthreads();
  mlp_layer<2, 8,  4>(wpk + O_DW3, p.db3, P0, PS, nullptr, 0, F0, nullptr, nullptr, nullptr, lane, wave); __syncthreads();
  // g0 . dW_1 -> GI  (GA is discarded scratch here)
  mlp_layer<0, 3, 16>(wpk + O_GW0, p.gb0, X,  XS, P0, PS, nullptr, nullptr, nullptr, nullptr, lane, wave); __syncthreads();
  mlp_layer<0, 8, 16>(wpk + O_GW1, p.gb1, P0, PS, P1, PS, nullptr, nullptr, nullptr, nullptr, lane, wave); __syncthreads();
  mlp_layer<0, 8, 16>(wpk + O_GW2, p.gb2, P1, PS, P0, PS, nullptr, nullptr, nullptr, nullptr, lane, wave); __syncthreads();
  mlp_layer<3, 8, 32>(wpk + O_GW3, p.gb3, P0, PS, nullptr, 0, GA, GI, DWA, DWA, lane, wave); __syncthreads();

  // ---- reversible-Heun time loop ------------------------------------------
  float *f0p = F0, *f1p = F1, *gip = GI, *gap = GA, *gbp = GB;
  float *dwa = DWA, *dwb = DWB;
  for (int n = 1; n < kT; ++n) {
    const float t1 = p.ts[n];
    // Phase A: z1 = 2y - z + f*dt + g.dW ; rebuild tx ; zero gdW ; stage dW_{n+1}
    for (int e = tid; e < TILE_M * kHid; e += 256) {
      const int m = e >> 6, h = e & 63;
      const float z1 = 2.0f * Y[m * FS + h] - Z[m * FS + h]
                     + f0p[m * FS + h] * dt + gip[m * FS + h];
      Z[m * FS + h] = z1;
      X[m * XS + 1 + h] = (_Float16)z1;
      gap[m * FS + h] = 0.0f;
      gbp[m * FS + h] = 0.0f;
    }
    for (int m = tid; m < TILE_M; m += 256) X[m * XS] = (_Float16)t1;
    for (int e = tid; e < TILE_M * kNoise; e += 256) {
      const int m = e >> 3, j = e & 7;
      dwb[e] = (n < kT - 1)
                 ? p.dW[((size_t)n * kBatch + (r0 + m)) * kNoise + j] * sdt
                 : 0.0f;
    }
    __syncthreads();

    // drift MLP -> f1
    mlp_layer<0, 3, 16>(wpk + O_DW0, p.db0, X,  XS, P0, PS, nullptr, nullptr, nullptr, nullptr, lane, wave); __syncthreads();
    mlp_layer<0, 8, 16>(wpk + O_DW1, p.db1, P0, PS, P1, PS, nullptr, nullptr, nullptr, nullptr, lane, wave); __syncthreads();
    mlp_layer<0, 8, 16>(wpk + O_DW2, p.db2, P1, PS, P0, PS, nullptr, nullptr, nullptr, nullptr, lane, wave); __syncthreads();
    mlp_layer<2, 8,  4>(wpk + O_DW3, p.db3, P0, PS, nullptr, 0, f1p, nullptr, nullptr, nullptr, lane, wave); __syncthreads();
    // diffusion MLP -> g1.dW_n (gap) and g1.dW_{n+1} (gbp)
    mlp_layer<0, 3, 16>(wpk + O_GW0, p.gb0, X,  XS, P0, PS, nullptr, nullptr, nullptr, nullptr, lane, wave); __syncthreads();
    mlp_layer<0, 8, 16>(wpk + O_GW1, p.gb1, P0, PS, P1, PS, nullptr, nullptr, nullptr, nullptr, lane, wave); __syncthreads();
    mlp_layer<0, 8, 16>(wpk + O_GW2, p.gb2, P1, PS, P0, PS, nullptr, nullptr, nullptr, nullptr, lane, wave); __syncthreads();
    mlp_layer<3, 8, 32>(wpk + O_GW3, p.gb3, P0, PS, nullptr, 0, gap, gbp, dwa, dwb, lane, wave); __syncthreads();

    // Phase B: y1 = y + dt/2 (f0 + f1) + 1/2 (g_prev + g_new).dW_n
    for (int e = tid; e < TILE_M * kHid; e += 256) {
      const int m = e >> 6, h = e & 63;
      Y[m * FS + h] += 0.5f * dt * (f0p[m * FS + h] + f1p[m * FS + h])
                     + 0.5f * (gip[m * FS + h] + gap[m * FS + h]);
    }
    __syncthreads();
    readout_step(Y, p.out, r0, n, t1, RWL, RBL, tid);
    __syncthreads();

    // rotate carries
    float* tmp;
    tmp = f0p; f0p = f1p; f1p = tmp;
    tmp = gip; gip = gbp; gbp = tmp;   // g_new.dW_{n+1} becomes next g_prev.dW
    tmp = dwa; dwa = dwb; dwb = tmp;
  }
}

}  // namespace

extern "C" void kernel_launch(void* const* d_in, const int* in_sizes, int n_in,
                              void* d_out, int out_size, void* d_ws, size_t ws_size,
                              hipStream_t stream) {
  (void)in_sizes; (void)n_in; (void)out_size; (void)ws_size;
  _Float16* wpk = (_Float16*)d_ws;

  struct Mat { const float* w; int K, N, KBT, off; };
  const Mat mats[12] = {
    { (const float*)d_in[4],   32, 256, 1, O_IW0 },
    { (const float*)d_in[6],  256, 256, 8, O_IW1 },
    { (const float*)d_in[8],  256, 256, 8, O_IW2 },
    { (const float*)d_in[10], 256,  64, 8, O_IW3 },
    { (const float*)d_in[12],  65, 256, 3, O_DW0 },
    { (const float*)d_in[14], 256, 256, 8, O_DW1 },
    { (const float*)d_in[16], 256, 256, 8, O_DW2 },
    { (const float*)d_in[18], 256,  64, 8, O_DW3 },
    { (const float*)d_in[20],  65, 256, 3, O_GW0 },
    { (const float*)d_in[22], 256, 256, 8, O_GW1 },
    { (const float*)d_in[24], 256, 256, 8, O_GW2 },
    { (const float*)d_in[26], 256, 512, 8, O_GW3 },
  };
  for (int i = 0; i < 12; ++i) {
    const int total  = mats[i].KBT * (mats[i].N >> 4) * 512;
    const int blocks = (total + 255) / 256;
    pack_weights<<<blocks, 256, 0, stream>>>(mats[i].w, wpk + mats[i].off,
                                             mats[i].K, mats[i].N, mats[i].KBT);
  }

  GenParams p;
  p.ts         = (const float*)d_in[0];
  p.init_noise = (const float*)d_in[2];
  p.dW         = (const float*)d_in[3];
  p.ib0 = (const float*)d_in[5];  p.ib1 = (const float*)d_in[7];
  p.ib2 = (const float*)d_in[9];  p.ib3 = (const float*)d_in[11];
  p.db0 = (const float*)d_in[13]; p.db1 = (const float*)d_in[15];
  p.db2 = (const float*)d_in[17]; p.db3 = (const float*)d_in[19];
  p.gb0 = (const float*)d_in[21]; p.gb1 = (const float*)d_in[23];
  p.gb2 = (const float*)d_in[25]; p.gb3 = (const float*)d_in[27];
  p.rW  = (const float*)d_in[28]; p.rb  = (const float*)d_in[29];
  p.wpk = wpk;
  p.out = (float*)d_out;

  sde_generator<<<dim3(kBatch / TILE_M), dim3(256), LDS_BYTES, stream>>>(p);
}
